// MultiHeadAttention_35021163331894
// MI455X (gfx1250) — matrix operand order repaired and checked by reading, not compiled, not run on hardware
//
#include <hip/hip_runtime.h>
#include <hip/hip_bf16.h>

// ---------------------------------------------------------------------------
// Fused multi-head attention for MI455X (gfx1250, wave32, WMMA 16x16x32 f16)
// B=2, S=2048, H=768, HEADS=12, D=64.  softmax over HEAD axis (faithful).
// mask is (B,1,S,S) broadcast over heads -> invariant under head-softmax -> skipped.
// All operands pre-converted to f16 so GEMM loops issue pure b128 loads with
// no cvt chains (keeps loads pipelined under the WMMAs).
// ---------------------------------------------------------------------------

typedef _Float16 f16;
typedef __attribute__((ext_vector_type(16))) _Float16 v16h;
typedef __attribute__((ext_vector_type(8)))  _Float16 v8h;
typedef __attribute__((ext_vector_type(4)))  _Float16 v4h;
typedef __attribute__((ext_vector_type(8)))  float    v8f;
typedef __attribute__((ext_vector_type(4)))  float    v4f;

#define B_     2
#define S_     2048
#define H_     768
#define HEADS_ 12
#define D_     64
#define X_ELEMS   (B_ * S_ * H_)            // 3,145,728 (also == QKV elems)
#define W_ELEMS   (H_ * H_)                 // 589,824
#define QKV_ELEMS (B_ * HEADS_ * S_ * D_)   // 3,145,728

// ---------------- f32 -> f16 pre-conversion (one-shot, bandwidth-trivial) ---
__global__ __launch_bounds__(256)
void cvt_f32_to_f16_kernel(const float* __restrict__ in, f16* __restrict__ out,
                           int n4) {
    const int i = blockIdx.x * blockDim.x + threadIdx.x;
    if (i < n4) {
        const v4f v = ((const v4f*)in)[i];
        v4h h;
#pragma unroll
        for (int j = 0; j < 4; ++j) h[j] = (f16)v[j];
        ((v4h*)out)[i] = h;
    }
}

// ---------------- WMMA fragment loaders (wave32, contiguous K order k = 16*(lane/16) + s, same for A and B) --------

// A-matrix 16x32 f16: lanes 0-15 row m=lane, K=0..15; lanes 16-31 row
// m=lane-16, K=16..31.  `p` points at row start (32 consecutive K vals).
__device__ __forceinline__ v16h load_a_f16(const f16* p, int lane) {
    const int ko = (lane & 16) ? 16 : 0;
    v8h lo = *(const v8h*)(p + ko);
    v8h hi = *(const v8h*)(p + ko + 8);
    v16h a;
#pragma unroll
    for (int i = 0; i < 8; ++i) { a[i] = lo[i]; a[8 + i] = hi[i]; }
    return a;
}

// B-matrix 32x16 f16: lanes 0-15 hold K=0..15 of column n=lane, lanes 16-31
// hold K=16..31 of column n=lane-16.  Caller passes the per-lane address of
// 16 consecutive K values (32B -> two independent global_load_b128).
__device__ __forceinline__ v16h load_b_f16(const f16* p) {
    return *(const v16h*)p;
}

__device__ __forceinline__ v8f wmma16x16x32(v16h a, v16h b, v8f c) {
    return __builtin_amdgcn_wmma_f32_16x16x32_f16(false, a, false, b,
                                                  (short)0, c, false, false);
}

// ---------------- Kernel 1: projection  y = x @ W^T + b  -> f16 -------------
// x: [B*S, H] f16 row-major, w: [H, H] f16 row-major (y[.,o] uses w[o, i]),
// out layout: transposed==0 -> [B, HEADS, S, D], transposed==1 -> [B, HEADS, D, S]
// Grid: (B*S)/16 blocks of 256 threads (8 waves); wave w computes N-tiles
// o_tile = w*6 + j, j=0..5 (48 tiles of 16 columns = 768).
__global__ __launch_bounds__(256)
void mha_proj_kernel(const f16* __restrict__ x, const f16* __restrict__ w,
                     const float* __restrict__ bias, f16* __restrict__ out,
                     int transposed) {
    const int lane   = threadIdx.x & 31;
    const int wave   = threadIdx.x >> 5;
    const int m_base = blockIdx.x * 16;
    const int m_row  = m_base + (lane & 15);        // A row for this lane

    v8f acc[6];
#pragma unroll
    for (int j = 0; j < 6; ++j) acc[j] = (v8f){0.f,0.f,0.f,0.f,0.f,0.f,0.f,0.f};

    for (int kk = 0; kk < H_; kk += 32) {
        const v16h a = load_a_f16(x + (size_t)m_row * H_ + kk, lane);
        const int ib = kk + ((lane & 16) ? 16 : 0);
        // Load all six B fragments first so the loads pipeline, then WMMA.
        v16h bf[6];
#pragma unroll
        for (int j = 0; j < 6; ++j) {
            const int o = (wave * 6 + j) * 16 + (lane & 15);
            bf[j] = load_b_f16(w + (size_t)o * H_ + ib);
        }
#pragma unroll
        for (int j = 0; j < 6; ++j) acc[j] = wmma16x16x32(a, bf[j], acc[j]);
    }

    // Add bias, convert to f16, scatter to [b,h,s,d] (or [b,h,d,s]).
#pragma unroll
    for (int j = 0; j < 6; ++j) {
        const int o    = (wave * 6 + j) * 16 + (lane & 15);
        const float bv = bias[o];
        const int head = o >> 6;
        const int d    = o & 63;
#pragma unroll
        for (int r = 0; r < 8; ++r) {
            const int m = m_base + r + ((lane & 16) ? 8 : 0);
            const int b = m >> 11;          // / S_
            const int s = m & (S_ - 1);
            const size_t idx = transposed
                ? ((size_t)(b * HEADS_ + head) * D_ + d) * S_ + s
                : ((size_t)(b * HEADS_ + head) * S_ + s) * D_ + d;
            out[idx] = (f16)(acc[j][r] + bv);
        }
    }
}

// ---------------- Kernel 2: fused attention + output projection -------------
// One block = 12 waves (one per head) x 32 lanes, handles a 16-row q tile.
// Loop over k in 32-wide tiles: scores (WMMA) -> LDS -> softmax across the
// 12 heads -> probs f16 in LDS -> P@V (WMMA).  Then staged context in LDS and
// output projection @ wo^T + bo (WMMA), f32 result to d_out.
__global__ __launch_bounds__(HEADS_ * 32)
void mha_attn_kernel(const f16* __restrict__ qh,   // [B,HEADS,S,D]
                     const f16* __restrict__ kh,   // [B,HEADS,S,D]
                     const f16* __restrict__ vT,   // [B,HEADS,D,S]
                     const f16* __restrict__ wo,   // [H,H] f16
                     const float* __restrict__ bo, // [H]
                     float* __restrict__ outp) {   // [B,S,H]
    __shared__ __align__(32) float sc[HEADS_][16][32];   // 24 KB
    __shared__ __align__(32) f16   pr[HEADS_][16][32];   // 12 KB
    __shared__ __align__(32) f16   ctx[16][H_];          // 24 KB

    const int lane   = threadIdx.x & 31;
    const int head   = threadIdx.x >> 5;
    const int b      = blockIdx.x >> 7;           // / (S_/16)
    const int q_base = (blockIdx.x & 127) * 16;

    const f16* qptr = qh + (size_t)(b * HEADS_ + head) * S_ * D_;
    const f16* kptr = kh + (size_t)(b * HEADS_ + head) * S_ * D_;
    const f16* vptr = vT + (size_t)(b * HEADS_ + head) * D_ * S_;

    // Q fragments for this head's 16 q rows, d-chunks 0..31 and 32..63.
    const int qrow = q_base + (lane & 15);
    const v16h qa0 = load_a_f16(qptr + (size_t)qrow * D_ + 0,  lane);
    const v16h qa1 = load_a_f16(qptr + (size_t)qrow * D_ + 32, lane);

    v8f oacc[4];
#pragma unroll
    for (int t = 0; t < 4; ++t) oacc[t] = (v8f){0.f,0.f,0.f,0.f,0.f,0.f,0.f,0.f};

    const float scale = 0.125f;   // 1/sqrt(64)
    const int n    = lane & 15;
    const int ko16 = (lane & 16) ? 16 : 0;
    const int m_of = (lane & 16) ? 8 : 0;

    for (int k_base = 0; k_base < S_; k_base += 32) {
        // Prefetch next K/V tiles toward the caches.
        if (k_base + 32 < S_) {
            __builtin_prefetch(kptr + (size_t)(k_base + 32 + n) * D_, 0, 1);
            __builtin_prefetch(vptr + (size_t)n * S_ + k_base + 32, 0, 1);
        }

        // ---- scores: 16q x 32k = Q(16x64) x K^T(64x32), two 16-wide n tiles
        v8f s0 = (v8f){0.f,0.f,0.f,0.f,0.f,0.f,0.f,0.f};
        v8f s1 = s0;
        {
            const f16* kr0 = kptr + (size_t)(k_base + n)      * D_;
            const f16* kr1 = kptr + (size_t)(k_base + 16 + n) * D_;
            // Issue all four B-fragment loads, then the four WMMAs.
            const v16h b00 = load_b_f16(kr0 + ko16);
            const v16h b10 = load_b_f16(kr1 + ko16);
            const v16h b01 = load_b_f16(kr0 + 32 + ko16);
            const v16h b11 = load_b_f16(kr1 + 32 + ko16);
            s0 = wmma16x16x32(qa0, b00, s0);
            s1 = wmma16x16x32(qa0, b10, s1);
            s0 = wmma16x16x32(qa1, b01, s0);
            s1 = wmma16x16x32(qa1, b11, s1);
        }
        // Scatter C layout into LDS (m = r + m_of, ncol = lane&15).
#pragma unroll
        for (int r = 0; r < 8; ++r) {
            sc[head][r + m_of][n]      = s0[r] * scale;
            sc[head][r + m_of][16 + n] = s1[r] * scale;
        }
        __syncthreads();

        // ---- softmax across the 12 heads for each of 16*32 (q,k) cells.
        for (int e = threadIdx.x; e < 16 * 32; e += HEADS_ * 32) {
            const int qq = e >> 5, kc = e & 31;
            float mx = sc[0][qq][kc];
#pragma unroll
            for (int h = 1; h < HEADS_; ++h) mx = fmaxf(mx, sc[h][qq][kc]);
            float sum = 0.f;
#pragma unroll
            for (int h = 0; h < HEADS_; ++h) {
                const float ex = __expf(sc[h][qq][kc] - mx);
                sc[h][qq][kc] = ex;
                sum += ex;
            }
            const float rs = __frcp_rn(sum);
#pragma unroll
            for (int h = 0; h < HEADS_; ++h)
                pr[h][qq][kc] = (f16)(sc[h][qq][kc] * rs);
        }
        __syncthreads();

        // ---- out += P(16x32) @ V(32x16) for four d tiles.
        const v16h pa = load_a_f16(&pr[head][lane & 15][0], lane);
        v16h vb[4];
#pragma unroll
        for (int dt = 0; dt < 4; ++dt)
            vb[dt] = load_b_f16(vptr + (size_t)(dt * 16 + n) * S_ + k_base + ko16);
#pragma unroll
        for (int dt = 0; dt < 4; ++dt)
            oacc[dt] = wmma16x16x32(pa, vb[dt], oacc[dt]);
        __syncthreads();   // protect sc/pr before next iteration's writes
    }

    // ---- stage context [16 x 768] f16 in LDS (per-head 64 columns).
#pragma unroll
    for (int dt = 0; dt < 4; ++dt)
#pragma unroll
        for (int r = 0; r < 8; ++r)
            ctx[r + m_of][head * D_ + dt * 16 + n] = (f16)oacc[dt][r];
    __syncthreads();

    // ---- output projection: y[16 x 768] = ctx @ wo^T + bo.
    // 48 N tiles over 12 waves -> 4 tiles per wave.
    v8f yacc[4];
#pragma unroll
    for (int j = 0; j < 4; ++j) yacc[j] = (v8f){0.f,0.f,0.f,0.f,0.f,0.f,0.f,0.f};

    for (int ii = 0; ii < H_; ii += 32) {
        const v16h a = load_a_f16(&ctx[lane & 15][ii], lane);
        v16h bf[4];
#pragma unroll
        for (int j = 0; j < 4; ++j) {
            const int o = (head * 4 + j) * 16 + n;
            bf[j] = load_b_f16(wo + (size_t)o * H_ + ii + ko16);
        }
#pragma unroll
        for (int j = 0; j < 4; ++j) yacc[j] = wmma16x16x32(a, bf[j], yacc[j]);
    }
#pragma unroll
    for (int j = 0; j < 4; ++j) {
        const int o    = (head * 4 + j) * 16 + n;
        const float bv = bo[o];
#pragma unroll
        for (int r = 0; r < 8; ++r) {
            const int m = q_base + r + m_of;
            outp[((size_t)b * S_ + m) * H_ + o] = yacc[j][r] + bv;
        }
    }
}

// ---------------------------------------------------------------------------
extern "C" void kernel_launch(void* const* d_in, const int* in_sizes, int n_in,
                              void* d_out, int out_size, void* d_ws, size_t ws_size,
                              hipStream_t stream) {
    const float* key   = (const float*)d_in[0];
    const float* value = (const float*)d_in[1];
    const float* query = (const float*)d_in[2];
    // d_in[3] = mask: broadcast over heads -> no-op under head-axis softmax.
    const float* wq = (const float*)d_in[4];
    const float* bq = (const float*)d_in[5];
    const float* wk = (const float*)d_in[6];
    const float* bk = (const float*)d_in[7];
    const float* wv = (const float*)d_in[8];
    const float* bv = (const float*)d_in[9];
    const float* wo = (const float*)d_in[10];
    const float* bo = (const float*)d_in[11];

    // Workspace layout (all f16):
    f16* qh   = (f16*)d_ws;                 // [B,HEADS,S,D]
    f16* kh   = qh  + QKV_ELEMS;            // [B,HEADS,S,D]
    f16* vT   = kh  + QKV_ELEMS;            // [B,HEADS,D,S]
    f16* xq   = vT  + QKV_ELEMS;            // query f16
    f16* xk   = xq  + X_ELEMS;              // key   f16
    f16* xv   = xk  + X_ELEMS;              // value f16
    f16* wq16 = xv  + X_ELEMS;
    f16* wk16 = wq16 + W_ELEMS;
    f16* wv16 = wk16 + W_ELEMS;
    f16* wo16 = wv16 + W_ELEMS;

    // One-shot f32 -> f16 conversions (activations + weights).
    {
        const int n4x = X_ELEMS / 4, n4w = W_ELEMS / 4;
        const int bx = (n4x + 255) / 256, bw = (n4w + 255) / 256;
        cvt_f32_to_f16_kernel<<<bx, 256, 0, stream>>>(query, xq, n4x);
        cvt_f32_to_f16_kernel<<<bx, 256, 0, stream>>>(key,   xk, n4x);
        cvt_f32_to_f16_kernel<<<bx, 256, 0, stream>>>(value, xv, n4x);
        cvt_f32_to_f16_kernel<<<bw, 256, 0, stream>>>(wq, wq16, n4w);
        cvt_f32_to_f16_kernel<<<bw, 256, 0, stream>>>(wk, wk16, n4w);
        cvt_f32_to_f16_kernel<<<bw, 256, 0, stream>>>(wv, wv16, n4w);
        cvt_f32_to_f16_kernel<<<bw, 256, 0, stream>>>(wo, wo16, n4w);
    }

    const int proj_blocks = (B_ * S_) / 16;     // 256
    mha_proj_kernel<<<proj_blocks, 256, 0, stream>>>(xq, wq16, bq, qh, 0);
    mha_proj_kernel<<<proj_blocks, 256, 0, stream>>>(xk, wk16, bk, kh, 0);
    mha_proj_kernel<<<proj_blocks, 256, 0, stream>>>(xv, wv16, bv, vT, 1);

    const int attn_blocks = B_ * (S_ / 16);     // 256
    mha_attn_kernel<<<attn_blocks, HEADS_ * 32, 0, stream>>>(qh, kh, vT, wo16, bo,
                                                             (float*)d_out);
}
